// FaultGAT_7739531067781
// MI455X (gfx1250) — matrix operand order repaired and checked by reading, not compiled, not run on hardware
//
#include <hip/hip_runtime.h>
#include <hip/hip_bf16.h>
#include <math.h>

#define NNODES 50000
#define NEDGES 800000
#define NLOOP  (NEDGES + NNODES)   /* edges + self loops = 850000 */
#define NPAD   50016               /* NNODES padded to multiple of 32 */
#define NEG_SLOPE 0.2f

typedef __attribute__((ext_vector_type(16))) _Float16 v16h;
typedef __attribute__((ext_vector_type(8)))  float    v8f;

// ---------------------------------------------------------------- utilities
__device__ __forceinline__ void atomicMaxF(float* a, float v) {
    // works for mixed-sign floats with -inf initialization
    if (v >= 0.0f) atomicMax((int*)a, __float_as_int(v));
    else           atomicMin((unsigned int*)a, __float_as_uint(v));
}

__global__ void fill_f32(float* p, int n, float v) {
    int i = blockIdx.x * blockDim.x + threadIdx.x;
    if (i < n) p[i] = v;
}

// ---------------------------------------------------------------- kernel A
// xl = x @ W for both conv_forward and conv_upstream, plus per-head
// attention dots (alpha_src/alpha_dst). Block = 256 threads = 4 nodes x 64
// channels; each wave32 = one (node, head) -> exact shuffle reduction width.
__global__ void gat_lin_kernel(const float* __restrict__ x,
                               const float* __restrict__ Wf,
                               const float* __restrict__ asf,
                               const float* __restrict__ adf,
                               const float* __restrict__ Wu,
                               const float* __restrict__ asu,
                               const float* __restrict__ adu,
                               float* __restrict__ xlf, float* __restrict__ xlu,
                               float* __restrict__ alsf, float* __restrict__ aldf,
                               float* __restrict__ alsu, float* __restrict__ aldu) {
    int t    = threadIdx.x;
    int node = blockIdx.x * 4 + (t >> 6);
    int c    = t & 63;                 // channel 0..63, head = c>>5
    float x0 = x[node * 4 + 0], x1 = x[node * 4 + 1];
    float x2 = x[node * 4 + 2], x3 = x[node * 4 + 3];
    float vf = x0 * Wf[c] + x1 * Wf[64 + c] + x2 * Wf[128 + c] + x3 * Wf[192 + c];
    float vu = x0 * Wu[c] + x1 * Wu[64 + c] + x2 * Wu[128 + c] + x3 * Wu[192 + c];
    xlf[node * 64 + c] = vf;
    xlu[node * 64 + c] = vu;
    // a_src/a_dst are [2,32] row-major -> flat index == c
    float psf = vf * asf[c], pdf = vf * adf[c];
    float psu = vu * asu[c], pdu = vu * adu[c];
#pragma unroll
    for (int off = 16; off >= 1; off >>= 1) {
        psf += __shfl_xor(psf, off, 32);
        pdf += __shfl_xor(pdf, off, 32);
        psu += __shfl_xor(psu, off, 32);
        pdu += __shfl_xor(pdu, off, 32);
    }
    if ((t & 31) == 0) {
        int h = c >> 5;
        alsf[node * 2 + h] = psf;
        aldf[node * 2 + h] = pdf;
        alsu[node * 2 + h] = psu;
        aldu[node * 2 + h] = pdu;
    }
}

// ---------------------------------------------------------------- edge passes
__device__ __forceinline__ void edge_sd(const int* __restrict__ src,
                                        const int* __restrict__ dst,
                                        int e, int& s, int& d) {
    if (e < NEDGES) { s = src[e]; d = dst[e]; }
    else            { s = e - NEDGES; d = s; }   // self loop
}

template <int HEADS>
__global__ void edge_max_kernel(const int* __restrict__ src, const int* __restrict__ dst,
                                const float* __restrict__ als, const float* __restrict__ ald,
                                float* __restrict__ m) {
    int e = blockIdx.x * blockDim.x + threadIdx.x;
    if (e >= NLOOP) return;
    int s, d; edge_sd(src, dst, e, s, d);
#pragma unroll
    for (int h = 0; h < HEADS; ++h) {
        float v = als[s * HEADS + h] + ald[d * HEADS + h];
        v = v > 0.0f ? v : NEG_SLOPE * v;
        atomicMaxF(&m[d * HEADS + h], v);
    }
}

template <int HEADS>
__global__ void edge_exp_kernel(const int* __restrict__ src, const int* __restrict__ dst,
                                const float* __restrict__ als, const float* __restrict__ ald,
                                const float* __restrict__ m, float* __restrict__ den,
                                float* __restrict__ coef) {
    int e = blockIdx.x * blockDim.x + threadIdx.x;
    if (e >= NLOOP) return;
    int s, d; edge_sd(src, dst, e, s, d);
#pragma unroll
    for (int h = 0; h < HEADS; ++h) {
        float v = als[s * HEADS + h] + ald[d * HEADS + h];
        v = v > 0.0f ? v : NEG_SLOPE * v;
        float ex = expf(v - m[d * HEADS + h]);
        atomicAdd(&den[d * HEADS + h], ex);
        coef[e * HEADS + h] = ex;
    }
}

template <int HEADS>
__global__ void edge_div_kernel(const int* __restrict__ src, const int* __restrict__ dst,
                                const float* __restrict__ den, float* __restrict__ coef) {
    int e = blockIdx.x * blockDim.x + threadIdx.x;
    if (e >= NLOOP) return;
    int s, d; edge_sd(src, dst, e, s, d);
#pragma unroll
    for (int h = 0; h < HEADS; ++h)
        coef[e * HEADS + h] = coef[e * HEADS + h] / den[d * HEADS + h];
}

template <int HEADS, int CH>
__global__ void edge_agg_kernel(const int* __restrict__ src, const int* __restrict__ dst,
                                const float* __restrict__ xl, const float* __restrict__ coef,
                                float* __restrict__ out) {
    int idx = blockIdx.x * blockDim.x + threadIdx.x;
    int e = idx / CH, c = idx % CH;
    if (e >= NLOOP) return;
    int s, d; edge_sd(src, dst, e, s, d);
    int h = (CH == HEADS) ? c : (c / (CH / HEADS));
    atomicAdd(&out[d * CH + c], xl[s * CH + c] * coef[e * HEADS + h]);
}

// ---------------------------------------------------------------- h = concat(relu,relu) -> f16
__global__ void build_h_kernel(const float* __restrict__ outf, const float* __restrict__ outu,
                               const float* __restrict__ bf, const float* __restrict__ bu,
                               _Float16* __restrict__ hf16) {
    int idx = blockIdx.x * blockDim.x + threadIdx.x;
    if (idx >= NPAD * 128) return;
    int node = idx >> 7, k = idx & 127;
    float v = 0.0f;
    if (node < NNODES) {
        if (k < 64) v = outf[node * 64 + k] + bf[k];
        else        v = outu[node * 64 + (k - 64)] + bu[k - 64];
        v = v > 0.0f ? v : 0.0f;
    }
    hf16[idx] = (_Float16)v;
}

// Pre-swizzle W_fc [128,64] f32 into WMMA B-fragment order:
// Wfrag[kk][ct][lane][j] with K = kk*32 + (lane>=16 ? 8:0) + (j&7) + (j>=8 ? 16:0),
// n = ct*16 + (lane&15).  Each lane's 16 halves become contiguous so the
// WMMA kernel loads B as two coalesced global_load_b128 per k-step.
__global__ void build_wfrag_kernel(const float* __restrict__ W,
                                   _Float16* __restrict__ Wfrag) {
    int i = blockIdx.x * blockDim.x + threadIdx.x;   // 4*4*32*16 = 8192
    if (i >= 4 * 4 * 32 * 16) return;
    int j    = i & 15;
    int lane = (i >> 4) & 31;
    int ct   = (i >> 9) & 3;
    int kk   = (i >> 11) & 3;
    int K = kk * 32 + ((lane & 16) ? 8 : 0) + (j & 7) + ((j & 8) ? 16 : 0);
    int n = ct * 16 + (lane & 15);
    Wfrag[i] = (_Float16)W[K * 64 + n];
}

// ---------------------------------------------------------------- FC via WMMA
// h2[N,64] = relu(h[N,128] @ W_fc[128,64] + b_fc). Block = 256 = 8 waves;
// block covers 32 rows; each wave owns one 16x16 tile (2 row-tiles x 4 col-tiles).
// K=128 -> 4 x v_wmma_f32_16x16x32_f16.
__global__ void fc_wmma_kernel(const _Float16* __restrict__ A,      // [NPAD,128]
                               const _Float16* __restrict__ Wfrag,  // fragment-order B
                               const float* __restrict__ bias,      // [64]
                               float* __restrict__ C) {             // [NNODES,64]
    int lane = threadIdx.x & 31;
    int w    = threadIdx.x >> 5;     // 0..7
    int rt   = w >> 2, ct = w & 3;
    int row0 = blockIdx.x * 32 + rt * 16;
    int col0 = ct * 16;
    int arow = row0 + (lane & 15);
    int hiA  = (lane & 16) ? 8 : 0;  // A: K split per lane half (ISA 16-bit A 16x32 layout)

    v8f acc = {};
    union { v16h v; _Float16 h[16]; } af, bfr;
#pragma unroll
    for (int kk = 0; kk < 4; ++kk) {
        int kb = kk * 32;
        const _Float16* ap = A + (size_t)arow * 128 + kb + hiA;
#pragma unroll
        for (int j = 0; j < 8; ++j) {        // halves 0..7 -> K=kb+hiA+j ; 8..15 -> +16
            af.h[j]     = ap[j];
            af.h[8 + j] = ap[16 + j];
        }
        const _Float16* bp = Wfrag + (((kk * 4 + ct) * 32 + lane) << 4);
#pragma unroll
        for (int j = 0; j < 16; ++j)          // contiguous -> 2x b128, fully coalesced
            bfr.h[j] = bp[j];
        acc = __builtin_amdgcn_wmma_f32_16x16x32_f16(
            false, af.v, false, bfr.v, (short)0, acc, false, false);
    }
    float bval = bias[col0 + (lane & 15)];
#pragma unroll
    for (int v = 0; v < 8; ++v) {
        int m = ((lane & 16) ? 8 : 0) + v;    // C/D layout: VGPR v -> M = v (+8 upper half)
        int r = row0 + m;
        if (r < NNODES) {
            float o = acc[v] + bval;
            C[(size_t)r * 64 + col0 + (lane & 15)] = o > 0.0f ? o : 0.0f;
        }
    }
}

// ---------------------------------------------------------------- output conv prep
__global__ void out_prep_kernel(const float* __restrict__ h2, const float* __restrict__ Wo,
                                const float* __restrict__ aso, const float* __restrict__ ado,
                                float* __restrict__ xlo, float* __restrict__ als,
                                float* __restrict__ ald) {
    int i = blockIdx.x * blockDim.x + threadIdx.x;
    if (i >= NNODES) return;
    float acc = 0.0f;
#pragma unroll
    for (int k = 0; k < 64; ++k) acc += h2[(size_t)i * 64 + k] * Wo[k];
    xlo[i] = acc;
    als[i] = acc * aso[0];
    ald[i] = acc * ado[0];
}

__global__ void sigmoid_kernel(const float* __restrict__ outo, const float* __restrict__ bo,
                               float* __restrict__ y) {
    int i = blockIdx.x * blockDim.x + threadIdx.x;
    if (i >= NNODES) return;
    float v = outo[i] + bo[0];
    y[i] = 1.0f / (1.0f + expf(-v));
}

// ---------------------------------------------------------------- launch
extern "C" void kernel_launch(void* const* d_in, const int* in_sizes, int n_in,
                              void* d_out, int out_size, void* d_ws, size_t ws_size,
                              hipStream_t stream) {
    const float* x   = (const float*)d_in[0];
    const int*   ei  = (const int*)  d_in[1];
    const float* Wf  = (const float*)d_in[2];
    const float* asf = (const float*)d_in[3];
    const float* adf = (const float*)d_in[4];
    const float* bfv = (const float*)d_in[5];
    const float* Wu  = (const float*)d_in[6];
    const float* asu = (const float*)d_in[7];
    const float* adu = (const float*)d_in[8];
    const float* buv = (const float*)d_in[9];
    const float* Wfc = (const float*)d_in[10];
    const float* bfc = (const float*)d_in[11];
    const float* Wo  = (const float*)d_in[12];
    const float* aso = (const float*)d_in[13];
    const float* ado = (const float*)d_in[14];
    const float* bo  = (const float*)d_in[15];
    const int* src = ei;
    const int* dst = ei + NEDGES;

    // ---- carve workspace
    char* w = (char*)d_ws;
    size_t off = 0;
    auto carve = [&](size_t bytes) -> void* {
        void* p = w + off;
        off = (off + bytes + 255) & ~(size_t)255;
        return p;
    };
    float* xlf   = (float*)carve((size_t)NNODES * 64 * 4);
    float* xlu   = (float*)carve((size_t)NNODES * 64 * 4);
    float* outf  = (float*)carve((size_t)NNODES * 64 * 4);
    float* outu  = (float*)carve((size_t)NNODES * 64 * 4);
    float* alsf  = (float*)carve((size_t)NNODES * 2 * 4);
    float* aldf  = (float*)carve((size_t)NNODES * 2 * 4);
    float* alsu  = (float*)carve((size_t)NNODES * 2 * 4);
    float* aldu  = (float*)carve((size_t)NNODES * 2 * 4);
    float* mf    = (float*)carve((size_t)NNODES * 2 * 4);
    float* denf  = (float*)carve((size_t)NNODES * 2 * 4);
    float* mu    = (float*)carve((size_t)NNODES * 2 * 4);
    float* denu  = (float*)carve((size_t)NNODES * 2 * 4);
    float* coeff = (float*)carve((size_t)NLOOP * 2 * 4);
    float* coefu = (float*)carve((size_t)NLOOP * 2 * 4);
    _Float16* hf16  = (_Float16*)carve((size_t)NPAD * 128 * 2);
    _Float16* Wfrag = (_Float16*)carve((size_t)4 * 4 * 32 * 16 * 2);
    float* xlo   = (float*)carve((size_t)NNODES * 4);
    float* also_ = (float*)carve((size_t)NNODES * 4);
    float* aldo  = (float*)carve((size_t)NNODES * 4);
    float* mo    = (float*)carve((size_t)NNODES * 4);
    float* deno  = (float*)carve((size_t)NNODES * 4);
    float* outo  = (float*)carve((size_t)NNODES * 4);
    float* h2    = xlf;    // alias: xlf dead after edge aggregation
    float* coefo = coeff;  // alias: coeff dead after forward aggregation

    const float NEGINF = -INFINITY;
    const int TB = 256;
    int bN64 = (NNODES * 64 + TB - 1) / TB;
    int bN2  = (NNODES * 2 + TB - 1) / TB;
    int bN   = (NNODES + TB - 1) / TB;
    int bE   = (NLOOP + TB - 1) / TB;
    int bE64 = (NLOOP * 64 + TB - 1) / TB;     // 212500 blocks
    int bH   = (NPAD * 128 + TB - 1) / TB;

    // ---- init accumulators (every call: graph replay must be deterministic)
    fill_f32<<<bN64, TB, 0, stream>>>(outf, NNODES * 64, 0.0f);
    fill_f32<<<bN64, TB, 0, stream>>>(outu, NNODES * 64, 0.0f);
    fill_f32<<<bN2, TB, 0, stream>>>(denf, NNODES * 2, 0.0f);
    fill_f32<<<bN2, TB, 0, stream>>>(denu, NNODES * 2, 0.0f);
    fill_f32<<<bN2, TB, 0, stream>>>(mf, NNODES * 2, NEGINF);
    fill_f32<<<bN2, TB, 0, stream>>>(mu, NNODES * 2, NEGINF);
    fill_f32<<<bN, TB, 0, stream>>>(deno, NNODES, 0.0f);
    fill_f32<<<bN, TB, 0, stream>>>(mo, NNODES, NEGINF);
    fill_f32<<<bN, TB, 0, stream>>>(outo, NNODES, 0.0f);

    // ---- linear + attention coefficients for both directions
    gat_lin_kernel<<<NNODES / 4, TB, 0, stream>>>(x, Wf, asf, adf, Wu, asu, adu,
                                                  xlf, xlu, alsf, aldf, alsu, aldu);

    // ---- forward conv (src -> dst)
    edge_max_kernel<2><<<bE, TB, 0, stream>>>(src, dst, alsf, aldf, mf);
    edge_exp_kernel<2><<<bE, TB, 0, stream>>>(src, dst, alsf, aldf, mf, denf, coeff);
    edge_div_kernel<2><<<bE, TB, 0, stream>>>(src, dst, denf, coeff);
    edge_agg_kernel<2, 64><<<bE64, TB, 0, stream>>>(src, dst, xlf, coeff, outf);

    // ---- upstream conv (dst -> src: swap pointers)
    edge_max_kernel<2><<<bE, TB, 0, stream>>>(dst, src, alsu, aldu, mu);
    edge_exp_kernel<2><<<bE, TB, 0, stream>>>(dst, src, alsu, aldu, mu, denu, coefu);
    edge_div_kernel<2><<<bE, TB, 0, stream>>>(dst, src, denu, coefu);
    edge_agg_kernel<2, 64><<<bE64, TB, 0, stream>>>(dst, src, xlu, coefu, outu);

    // ---- h = concat(relu(outf+bf), relu(outu+bu)) as f16, zero-padded rows
    build_h_kernel<<<bH, TB, 0, stream>>>(outf, outu, bfv, buv, hf16);
    build_wfrag_kernel<<<(4 * 4 * 32 * 16 + TB - 1) / TB, TB, 0, stream>>>(Wfc, Wfrag);

    // ---- FC layer on tensor cores: h2 = relu(h @ W_fc + b_fc)
    fc_wmma_kernel<<<NPAD / 32, TB, 0, stream>>>(hf16, Wfrag, bfc, h2);

    // ---- output GATConv (1 head, 1 channel, concat=False)
    out_prep_kernel<<<bN, TB, 0, stream>>>(h2, Wo, aso, ado, xlo, also_, aldo);
    edge_max_kernel<1><<<bE, TB, 0, stream>>>(src, dst, also_, aldo, mo);
    edge_exp_kernel<1><<<bE, TB, 0, stream>>>(src, dst, also_, aldo, mo, deno, coefo);
    edge_div_kernel<1><<<bE, TB, 0, stream>>>(src, dst, deno, coefo);
    edge_agg_kernel<1, 1><<<bE, TB, 0, stream>>>(src, dst, xlo, coefo, outo);

    // ---- sigmoid output
    sigmoid_kernel<<<bN, TB, 0, stream>>>(outo, bo, (float*)d_out);
}
